// SpikingUnifiedCoreFlow_25383256719438
// MI455X (gfx1250) — matrix operand order repaired and checked by reading, not compiled, not run on hardware
//
#include <hip/hip_runtime.h>

typedef __attribute__((ext_vector_type(16))) _Float16 v16h;
typedef __attribute__((ext_vector_type(8)))  _Float16 v8h;
typedef __attribute__((ext_vector_type(8)))  float    v8f;

#define SIM_T 32
#define BATCH 64

// ---------------- conversion kernels ----------------
__global__ void cvt_f32_to_f16(const float* __restrict__ src,
                               _Float16* __restrict__ dst, int n) {
  int i = blockIdx.x * blockDim.x + threadIdx.x;
  if (i < n) dst[i] = (_Float16)src[i];
}

// w2 is (1000, 1024); pad to (1024, 1024) f16 with zero rows so the N
// dimension is a multiple of the 128-wide block tile.
__global__ void cvt_pad_w2(const float* __restrict__ w2,
                           _Float16* __restrict__ dst) {
  int i = blockIdx.x * blockDim.x + threadIdx.x;   // < 1024*1024
  int n = i >> 10;
  int k = i & 1023;
  dst[i] = (n < 1000) ? (_Float16)w2[n * 1024 + k] : (_Float16)0.0f;
}

// ---------------- uniform spike encoder (+ column gather) ----------------
// A[(t*BATCH + b) * D + k] = spike(t, rates[b, idx[k]])  as f16 {0,1}
__global__ void spike_gen(const float* __restrict__ rates, int rstride,
                          const int* __restrict__ idx,
                          _Float16* __restrict__ A, int D) {
  int k = blockIdx.x * blockDim.x + threadIdx.x;
  int b = blockIdx.y;
  if (k >= D) return;
  float r = rates[(size_t)b * rstride + idx[k]];
  int   N = (int)rintf(r * (float)SIM_T);          // round-half-even, like jnp.round
  int   Ns = N < 1 ? 1 : N;
  float spacing = (float)SIM_T / (float)Ns;
  size_t col = (size_t)b * D + k;
  #pragma unroll
  for (int t = 0; t < SIM_T; ++t) {
    float s;
    if (N >= SIM_T)      s = 1.0f;
    else if (N <= 0)     s = 0.0f;
    else {
      float ct = (float)t;
      bool fire = (floorf(ct / spacing) < (float)N) &&
                  (floorf(fmodf(ct, spacing)) == 0.0f);
      s = fire ? 1.0f : 0.0f;
    }
    A[(size_t)t * BATCH * D + col] = (_Float16)s;
  }
}

// ---------------- LDS-staged f16 WMMA GEMM: C[M,N] = A[M,K] * B[N,K]^T ----
// Block = 256 threads (8 waves). Block tile 128x128, K-chunk 64.
// Wave w computes rows [w*16, w*16+16) x all 128 block columns:
// 8 accumulators, A fragment reused 8x per WMMA K-step, B tile shared
// across all 8 waves via LDS (cuts L2 traffic ~4.5x vs streaming).
#define BM 128
#define BN 128
#define BK 64

__global__ void __launch_bounds__(256)
gemm_wmma_f16(const _Float16* __restrict__ A,
              const _Float16* __restrict__ B,
              float* __restrict__ C,
              int M, int N, int K) {
  __shared__ _Float16 As[BM * BK];   // [m_local][k_local], 16 KB
  __shared__ _Float16 Bs[BN * BK];   // [n_local][k_local], 16 KB

  const int tid  = threadIdx.x;
  const int lane = tid & 31;
  const int w    = tid >> 5;         // wave 0..7
  const int lr   = lane & 15;        // row (A) / col (B,C) within 16-tile
  const int lg   = lane >> 4;        // K-group selector

  const int mbase = blockIdx.y * BM; // rows of C
  const int nbase = blockIdx.x * BN; // cols of C
  (void)M; (void)N;

  // Global->LDS copy assignment: thread t moves one 64-byte row segment
  // of A and one of B per K-chunk (128 rows x 2 segments of 32 halves).
  const int cr = tid >> 1;           // 0..127 (row within tile)
  const int cc = (tid & 1) * 32;     // 0 or 32 (k offset within chunk)
  const _Float16* gA = A + (size_t)(mbase + cr) * K + cc;
  const _Float16* gB = B + (size_t)(nbase + cr) * K + cc;
  _Float16* sA = As + cr * BK + cc;
  _Float16* sB = Bs + cr * BK + cc;

  v8f acc[8];
  #pragma unroll
  for (int j = 0; j < 8; ++j) acc[j] = (v8f){};

  const _Float16* aRow = As + (w * 16 + lr) * BK;

  for (int k0 = 0; k0 < K; k0 += BK) {
    // ---- stage A/B tiles into LDS (global_load_b128 -> ds_store_b128) ----
    #pragma unroll
    for (int s = 0; s < 4; ++s) {
      *(v8h*)(sA + 8 * s) = *(const v8h*)(gA + k0 + 8 * s);
      *(v8h*)(sB + 8 * s) = *(const v8h*)(gB + k0 + 8 * s);
    }
    __builtin_prefetch(gA + k0 + BK, 0, 1);
    __builtin_prefetch(gB + k0 + BK, 0, 1);
    __syncthreads();

    // ---- two WMMA K-steps per chunk ----
    #pragma unroll
    for (int kk = 0; kk < BK; kk += 32) {
      // A fragment (ISA 16-bit A layout): two contiguous 16B chunks
      v8h alo = *(const v8h*)(aRow + kk + lg * 8);
      v8h ahi = *(const v8h*)(aRow + kk + 16 + lg * 8);
      v16h a  = __builtin_shufflevector(alo, ahi,
                  0,1,2,3,4,5,6,7,8,9,10,11,12,13,14,15);
      #pragma unroll
      for (int j = 0; j < 8; ++j) {
        // B fragment: col lr of block-column group j, same two 8-K chunks as A
        const _Float16* bRow = Bs + (j * 16 + lr) * BK;
        v8h blo = *(const v8h*)(bRow + kk + lg * 8);
        v8h bhi = *(const v8h*)(bRow + kk + 16 + lg * 8);
        v16h b  = __builtin_shufflevector(blo, bhi,
                    0,1,2,3,4,5,6,7,8,9,10,11,12,13,14,15);
        acc[j] = __builtin_amdgcn_wmma_f32_16x16x32_f16(
                     false, a, false, b, (short)0, acc[j], false, false);
      }
    }
    __syncthreads();
  }

  // ---- store D (ISA 16x16 f32 C/D layout) ----
  const int mrow = mbase + w * 16 + lg * 8;
  #pragma unroll
  for (int r = 0; r < 8; ++r) {
    float* Cr = C + (size_t)(mrow + r) * N + nbase + lr;
    #pragma unroll
    for (int j = 0; j < 8; ++j) Cr[j * 16] = acc[j][r];
  }
}

// ---------------- LIF membrane scan -> spike count ----------------
__global__ void lif_scan_count(const float* __restrict__ Z,
                               const float* __restrict__ thp,
                               float* __restrict__ cnt, int D) {
  int o = blockIdx.x * blockDim.x + threadIdx.x;
  int b = blockIdx.y;
  if (o >= D) return;
  const float th = thp[0];
  float memb = 0.0f, c = 0.0f;
  const float* p = Z + (size_t)b * D + o;
  #pragma unroll
  for (int t = 0; t < SIM_T; ++t) {
    memb += p[(size_t)t * BATCH * D];
    if (th < memb) { memb -= th; c += 1.0f; }
  }
  cnt[(size_t)b * D + o] = c;
}

// ---------------- gather(idx1) + avg_pool2d(2x2) + clip ----------------
__global__ void pool_rates(const float* __restrict__ cnt,
                           const int* __restrict__ idx1,
                           float* __restrict__ y) {
  int p = blockIdx.x * blockDim.x + threadIdx.x;   // 0..1023
  int b = blockIdx.y;
  int c = p >> 4;
  int i = (p >> 2) & 3;
  int j = p & 3;
  float s = 0.0f;
  #pragma unroll
  for (int dh = 0; dh < 2; ++dh)
    #pragma unroll
    for (int dw = 0; dw < 2; ++dw) {
      int h = 2 * i + dh, wd = 2 * j + dw;
      s += cnt[(size_t)b * 4096 + idx1[c * 64 + h * 8 + wd]];
    }
  float rate = s * (1.0f / (4.0f * (float)SIM_T));  // mean over T and over 2x2
  y[(size_t)b * 1024 + p] = fminf(fmaxf(rate, 0.0f), 1.0f);
}

// ---------------- final LIF scan + idx_out gather -> spike counts ----------
__global__ void lif_scan_out(const float* __restrict__ Z,
                             const float* __restrict__ thp,
                             const int* __restrict__ idx_out,
                             float* __restrict__ out) {
  int j = blockIdx.x * blockDim.x + threadIdx.x;   // output neuron index
  int b = blockIdx.y;
  if (j >= 1000) return;
  int o = idx_out[j];
  const float th = thp[0];
  float memb = 0.0f, c = 0.0f;
  const float* p = Z + (size_t)b * 1024 + o;       // Z stride padded to 1024
  #pragma unroll
  for (int t = 0; t < SIM_T; ++t) {
    memb += p[(size_t)t * BATCH * 1024];
    if (th < memb) { memb -= th; c += 1.0f; }
  }
  out[(size_t)b * 1000 + j] = c;
}

extern "C" void kernel_launch(void* const* d_in, const int* in_sizes, int n_in,
                              void* d_out, int out_size, void* d_ws, size_t ws_size,
                              hipStream_t stream) {
  (void)in_sizes; (void)n_in; (void)out_size; (void)ws_size;
  const float* x       = (const float*)d_in[0];   // (64, 4096)
  const float* w0      = (const float*)d_in[1];   // (4096, 4096)
  const float* t0      = (const float*)d_in[2];   // scalar
  const float* w2      = (const float*)d_in[3];   // (1000, 1024)
  const float* t2      = (const float*)d_in[4];   // scalar
  const int*   idx0    = (const int*)d_in[5];     // (4096,)
  const int*   idx1    = (const int*)d_in[6];     // (4096,)
  const int*   idx2    = (const int*)d_in[7];     // (1024,)
  const int*   idx_out = (const int*)d_in[8];     // (1000,)
  float* out = (float*)d_out;

  // Workspace layout (256B aligned); A1/Z2 reuse A0/Z0 regions.
  char* ws = (char*)d_ws;
  size_t off = 0;
  auto take = [&](size_t bytes) { size_t o = off; off += (bytes + 255) & ~(size_t)255; return o; };
  _Float16* w0h = (_Float16*)(ws + take((size_t)4096 * 4096 * 2)); // 32 MB
  _Float16* w2h = (_Float16*)(ws + take((size_t)1024 * 1024 * 2)); //  2 MB
  _Float16* A0  = (_Float16*)(ws + take((size_t)2048 * 4096 * 2)); // 16 MB (reused as A1)
  float*    Z0  = (float*)   (ws + take((size_t)2048 * 4096 * 4)); // 32 MB (reused as Z2)
  float*    cnt = (float*)   (ws + take((size_t)BATCH * 4096 * 4));//  1 MB
  float*    yr  = (float*)   (ws + take((size_t)BATCH * 1024 * 4));// 256 KB
  _Float16* A1  = A0;
  float*    Z2  = Z0;

  const int TB = 256;

  // 1) weights -> f16
  cvt_f32_to_f16<<<dim3((4096u * 4096u) / TB), dim3(TB), 0, stream>>>(w0, w0h, 4096 * 4096);
  cvt_pad_w2  <<<dim3((1024u * 1024u) / TB), dim3(TB), 0, stream>>>(w2, w2h);

  // 2) input spike train, gathered by idx0 -> A0 (2048 x 4096 f16)
  spike_gen<<<dim3(4096 / TB, BATCH), dim3(TB), 0, stream>>>(x, 4096, idx0, A0, 4096);

  // 3) core-0 batched GEMM: Z0 = A0 @ w0^T   (2048 x 4096, K = 4096)
  gemm_wmma_f16<<<dim3(4096 / BN, 2048 / BM), dim3(TB), 0, stream>>>(A0, w0h, Z0, 2048, 4096, 4096);

  // 4) LIF scan core 0 -> spike counts (64 x 4096)
  lif_scan_count<<<dim3(4096 / TB, BATCH), dim3(TB), 0, stream>>>(Z0, t0, cnt, 4096);

  // 5) gather idx1 + avg_pool2d + clip -> rates (64 x 1024)
  pool_rates<<<dim3(1024 / TB, BATCH), dim3(TB), 0, stream>>>(cnt, idx1, yr);

  // 6) second spike train, gathered by idx2 -> A1 (2048 x 1024 f16)
  spike_gen<<<dim3(1024 / TB, BATCH), dim3(TB), 0, stream>>>(yr, 1024, idx2, A1, 1024);

  // 7) core-2 batched GEMM: Z2 = A1 @ w2^T   (2048 x 1024pad, K = 1024)
  gemm_wmma_f16<<<dim3(1024 / BN, 2048 / BM), dim3(TB), 0, stream>>>(A1, w2h, Z2, 2048, 1024, 1024);

  // 8) LIF scan core 2 + idx_out gather -> output spike counts (64 x 1000)
  lif_scan_out<<<dim3((1000 + TB - 1) / TB, BATCH), dim3(TB), 0, stream>>>(Z2, t2, idx_out, out);
}
